// FFD_26242250179225
// MI455X (gfx1250) — compile-verified, matrix operand order checked by reading, not run on hardware
//
#include <hip/hip_runtime.h>
#include <hip/hip_bf16.h>

typedef __attribute__((ext_vector_type(8)))  float  v8f;
typedef __attribute__((ext_vector_type(8)))  __bf16 v8bf;
typedef __attribute__((ext_vector_type(16))) __bf16 v16bf;

#define Bsz 128
#define Tsz 254
#define Dsz 768
#define twoD 1536
#define twoH 2048
#define Hsz 1024
#define MTOT (Bsz * Tsz)      // 32512 = 2032 * 16 exactly
#define FSH 1544              // feats LDS stride (halfs): 1544*2=3088 B, %256==16 -> conflict-free b128
#define H1SH 2056             // h1 LDS stride (halfs): 2056*2=4112 B, %256==16 -> conflict-free b128
#define H2S 1028              // h2 LDS stride (f32)

static __device__ __forceinline__ v8f wmma_bf16(v16bf a, v16bf b, v8f c) {
  // D(16x16 f32) = A(16x32 bf16) * B(32x16 bf16) + C
  return __builtin_amdgcn_wmma_f32_16x16x32_bf16(
      /*neg_a=*/false, a, /*neg_b=*/false, b,
      /*c_mod=*/(short)0, c, /*reuse_a=*/false, /*reuse_b=*/false);
}

static __device__ __forceinline__ v16bf pack16(v8bf a, v8bf b) {
  return __builtin_shufflevector(a, b, 0, 1, 2, 3, 4, 5, 6, 7,
                                       8, 9, 10, 11, 12, 13, 14, 15);
}

// ---------------------------------------------------------------------------
// Kernel 1: exclusive prefix mean along T for each (b,d) lane (coalesced in d).
// ---------------------------------------------------------------------------
__global__ void ffd_prefix_mean_kernel(const float* __restrict__ h,
                                       float* __restrict__ P) {
  int idx = blockIdx.x * blockDim.x + threadIdx.x;   // b*D + d
  if (idx >= Bsz * Dsz) return;
  int b = idx / Dsz;
  int d = idx - b * Dsz;
  const float* hp = h + (size_t)b * Tsz * Dsz + d;
  float* pp = P + (size_t)b * Tsz * Dsz + d;
  float s = 0.0f;
  for (int t = 0; t < Tsz; ++t) {
    float inv = 1.0f / (float)(t > 0 ? t : 1);
    pp[(size_t)t * Dsz] = s * inv;
    s += hp[(size_t)t * Dsz];
  }
}

// ---------------------------------------------------------------------------
// Kernel 2: split W[K][N] (f32) into hi/lo bf16, pre-swizzled into the WMMA
// 32x16 B-fragment lane layout: chunk (kt, kg, n) holds 16 halfs, where
// half j maps to K = kt*32 + kg*8 + j (j<8) or kt*32 + kg*8 + j + 8 (j>=8).
// A lane then reads its whole fragment as one contiguous 32-byte chunk.
// ---------------------------------------------------------------------------
__global__ void ffd_prep_w_kernel(const float* __restrict__ W, int K, int N,
                                  __bf16* __restrict__ outH,
                                  __bf16* __restrict__ outL) {
  int idx = blockIdx.x * blockDim.x + threadIdx.x;
  int total = (K / 32) * 2 * N * 16;
  if (idx >= total) return;
  int j = idx & 15;
  int rest = idx >> 4;
  int n = rest % N;
  int kc = rest / N;                 // kt*2 + kg
  int kg = kc & 1;
  int kt = kc >> 1;
  int k = kt * 32 + kg * 8 + (j < 8 ? j : j + 8);
  float w = W[(size_t)k * N + n];
  __bf16 hi = (__bf16)w;
  outH[idx] = hi;
  outL[idx] = (__bf16)(w - (float)hi);
}

// ---------------------------------------------------------------------------
// Kernel 3: fused 3-layer MLP for a 16-token tile; GEMM1/2 via split-bf16
// (hi*hi + hi*lo + lo*hi) on v_wmma_f32_16x16x32_bf16. h1/h2 live in LDS.
// ---------------------------------------------------------------------------
__global__ __launch_bounds__(256) void ffd_mlp_fused_kernel(
    const float* __restrict__ P, const float* __restrict__ Hs,
    const __bf16* __restrict__ W1H, const __bf16* __restrict__ W1L,
    const float* __restrict__ b1,
    const __bf16* __restrict__ W2H, const __bf16* __restrict__ W2L,
    const float* __restrict__ b2,
    const float* __restrict__ W3, const float* __restrict__ b3,
    float* __restrict__ out) {
  extern __shared__ char smemc[];
  __bf16* fH  = (__bf16*)smemc;                       // 16 x FSH  (49,408 B)
  __bf16* fLo = fH + 16 * FSH;                        // 16 x FSH  (-> 98,816 B)
  __bf16* h1H = (__bf16*)(smemc + 2 * 16 * FSH * 2);  // 16 x H1SH (65,792 B)
  __bf16* h1L = h1H + 16 * H1SH;                      // 16 x H1SH (-> 230,400 B)
  float* h2L  = (float*)smemc;                        // reuses feats region
  float* partL = (float*)smemc + 16 * H2S;            // 16x16 partials

  const int tid  = threadIdx.x;
  const int wave = tid >> 5;
  const int lane = tid & 31;
  const int m0   = blockIdx.x * 16;

  const int ln15 = lane & 15;        // A row / B col / D col
  const int kg   = lane >> 4;        // half-wave K-group select
  const int kg8  = kg * 8;
  const int rb   = kg * 8;           // D row base

  // ---- stage feats tile [16 x 1536] into LDS as hi/lo bf16 (coalesced) ----
  for (int i = tid; i < 16 * twoD; i += 256) {
    int r = i / twoD;
    int k = i - r * twoD;
    size_t g = (size_t)(m0 + r) * Dsz;
    float v = (k < Dsz) ? P[g + k] : Hs[g + (k - Dsz)];
    __bf16 hi = (__bf16)v;
    fH[r * FSH + k]  = hi;
    fLo[r * FSH + k] = (__bf16)(v - (float)hi);
  }
  __syncthreads();

  // ---- GEMM1: feats[16x1536] @ W1[1536x2048], bias+relu -> h1 hi/lo ----
  {
    const int nw = wave * 256;
    for (int g = 0; g < 4; ++g) {
      const int ng = nw + g * 64;
      v8f acc[4] = {};
#pragma unroll 1
      for (int kt = 0; kt < twoD / 32; ++kt) {
        const int kk = kt * 32;
        const __bf16* pa = fH + ln15 * FSH + kk + kg8;
        v16bf Ah = pack16(*(const v8bf*)pa, *(const v8bf*)(pa + 16));
        const __bf16* pl = fLo + ln15 * FSH + kk + kg8;
        v16bf Al = pack16(*(const v8bf*)pl, *(const v8bf*)(pl + 16));
        const size_t boff = ((size_t)(kt * 2 + kg) * twoH + ng + ln15) * 16;
#pragma unroll
        for (int j = 0; j < 4; ++j) {
          v16bf Bh = *(const v16bf*)(W1H + boff + (size_t)j * 256);
          v16bf Bl = *(const v16bf*)(W1L + boff + (size_t)j * 256);
          acc[j] = wmma_bf16(Ah, Bh, acc[j]);
          acc[j] = wmma_bf16(Ah, Bl, acc[j]);
          acc[j] = wmma_bf16(Al, Bh, acc[j]);
        }
      }
#pragma unroll
      for (int j = 0; j < 4; ++j) {
        const int n = ng + j * 16 + ln15;
        const float bias = b1[n];
#pragma unroll
        for (int i = 0; i < 8; ++i) {
          float v = acc[j][i] + bias;
          v = v > 0.0f ? v : 0.0f;
          __bf16 hi = (__bf16)v;
          h1H[(rb + i) * H1SH + n] = hi;
          h1L[(rb + i) * H1SH + n] = (__bf16)(v - (float)hi);
        }
      }
    }
  }
  __syncthreads();

  // ---- GEMM2: h1[16x2048] @ W2[2048x1024], bias+relu -> h2 (f32 LDS) ----
  {
    const int nw = wave * 128;
    for (int g = 0; g < 2; ++g) {
      const int ng = nw + g * 64;
      v8f acc[4] = {};
#pragma unroll 1
      for (int kt = 0; kt < twoH / 32; ++kt) {
        const int kk = kt * 32;
        const __bf16* pa = h1H + ln15 * H1SH + kk + kg8;
        v16bf Ah = pack16(*(const v8bf*)pa, *(const v8bf*)(pa + 16));
        const __bf16* pl = h1L + ln15 * H1SH + kk + kg8;
        v16bf Al = pack16(*(const v8bf*)pl, *(const v8bf*)(pl + 16));
        const size_t boff = ((size_t)(kt * 2 + kg) * Hsz + ng + ln15) * 16;
#pragma unroll
        for (int j = 0; j < 4; ++j) {
          v16bf Bh = *(const v16bf*)(W2H + boff + (size_t)j * 256);
          v16bf Bl = *(const v16bf*)(W2L + boff + (size_t)j * 256);
          acc[j] = wmma_bf16(Ah, Bh, acc[j]);
          acc[j] = wmma_bf16(Ah, Bl, acc[j]);
          acc[j] = wmma_bf16(Al, Bh, acc[j]);
        }
      }
#pragma unroll
      for (int j = 0; j < 4; ++j) {
        const int n = ng + j * 16 + ln15;
        const float bias = b2[n];
#pragma unroll
        for (int i = 0; i < 8; ++i) {
          float v = acc[j][i] + bias;
          h2L[(rb + i) * H2S + n] = v > 0.0f ? v : 0.0f;
        }
      }
    }
  }
  __syncthreads();

  // ---- GEMM3: h2[16x1024] @ W3[1024x1] + b3 -> out ----
  {
    const int m = tid & 15;
    const int hs = tid >> 4;
    float s = 0.0f;
#pragma unroll 4
    for (int h = hs; h < Hsz; h += 16) s += h2L[m * H2S + h] * W3[h];
    partL[hs * 16 + m] = s;
  }
  __syncthreads();
  if (tid < 16) {
    float acc = b3[0];
#pragma unroll
    for (int p = 0; p < 16; ++p) acc += partL[p * 16 + tid];
    out[m0 + tid] = acc;
  }
}

extern "C" void kernel_launch(void* const* d_in, const int* in_sizes, int n_in,
                              void* d_out, int out_size, void* d_ws, size_t ws_size,
                              hipStream_t stream) {
  const float* h  = (const float*)d_in[0];
  const float* W1 = (const float*)d_in[1];
  const float* b1 = (const float*)d_in[2];
  const float* W2 = (const float*)d_in[3];
  const float* b2 = (const float*)d_in[4];
  const float* W3 = (const float*)d_in[5];
  const float* b3 = (const float*)d_in[6];
  float* out = (float*)d_out;

  // Workspace layout (bytes):
  //   P      : [B,T,D] f32 prefix means                  99,876,864
  //   W1H/W1L: swizzled bf16 hi/lo of W1   2 x 6,291,456
  //   W2H/W2L: swizzled bf16 hi/lo of W2   2 x 4,194,304
  char* ws = (char*)d_ws;
  float* P = (float*)ws;
  size_t off = (size_t)MTOT * Dsz * sizeof(float);
  __bf16* W1Hs = (__bf16*)(ws + off); off += (size_t)twoD * twoH * 2;
  __bf16* W1Ls = (__bf16*)(ws + off); off += (size_t)twoD * twoH * 2;
  __bf16* W2Hs = (__bf16*)(ws + off); off += (size_t)twoH * Hsz * 2;
  __bf16* W2Ls = (__bf16*)(ws + off);

  ffd_prefix_mean_kernel<<<(Bsz * Dsz + 255) / 256, 256, 0, stream>>>(h, P);

  int tw1 = (twoD / 32) * 2 * twoH * 16;   // 3,145,728
  int tw2 = (twoH / 32) * 2 * Hsz * 16;    // 2,097,152
  ffd_prep_w_kernel<<<(tw1 + 255) / 256, 256, 0, stream>>>(W1, twoD, twoH, W1Hs, W1Ls);
  ffd_prep_w_kernel<<<(tw2 + 255) / 256, 256, 0, stream>>>(W2, twoH, Hsz, W2Hs, W2Ls);

  size_t lds_bytes = (size_t)(2 * 16 * FSH + 2 * 16 * H1SH) * sizeof(__bf16); // 230,400 B
  ffd_mlp_fused_kernel<<<MTOT / 16, 256, lds_bytes, stream>>>(
      P, h, W1Hs, W1Ls, b1, W2Hs, W2Ls, b2, W3, b3, out);
}
